// PraxisScatter_84439057039460
// MI455X (gfx1250) — matrix operand order repaired and checked by reading, not compiled, not run on hardware
//
#include <hip/hip_runtime.h>
#include <math.h>

#ifndef __has_builtin
#define __has_builtin(x) 0
#endif

// ---------------------------------------------------------------------------
// Types for CDNA5 WMMA (wave32)
// ---------------------------------------------------------------------------
typedef __bf16 bf16;
typedef __attribute__((ext_vector_type(16))) __bf16 v16bf;
typedef __attribute__((ext_vector_type(8)))  __bf16 v8bf;
typedef __attribute__((ext_vector_type(8)))  float  v8f;
typedef __attribute__((ext_vector_type(4)))  unsigned int v4u;
typedef __attribute__((ext_vector_type(8)))  int v8i;
typedef __attribute__((ext_vector_type(4)))  int v4i;

#if defined(__gfx1250__) && __has_builtin(__builtin_amdgcn_tensor_load_to_lds) && \
    __has_builtin(__builtin_amdgcn_s_wait_tensorcnt)
#define USE_TDM 1
#else
#define USE_TDM 0
#endif

static __device__ inline v16bf ld16(const bf16* p0, const bf16* p1) {
    v8bf lo = *(const v8bf*)p0;
    v8bf hi = *(const v8bf*)p1;
    v16bf r;
#pragma unroll
    for (int i = 0; i < 8; ++i) { r[i] = lo[i]; r[i + 8] = hi[i]; }
    return r;
}

#if USE_TDM
// Issue one TDM 2D tile load: 64 rows x 32 bf16 from row-major [rowsTot x K]
// global (row pitch K elements) into LDS at byte offset ldsOff, with LDS
// padding of 4 DWORDs after every 16 DWORDs (row pitch 64B -> 80B).
static __device__ inline void tdm_issue_a(const bf16* gsrc, unsigned ldsOff,
                                          int K, int rowsTot) {
    unsigned long long ga = (unsigned long long)(uintptr_t)gsrc;
    v4u g0;
    g0[0] = 1u;                                              // count=1, user mode
    g0[1] = ldsOff;                                          // lds_addr (bytes)
    g0[2] = (unsigned)ga;                                    // global_addr lo
    g0[3] = (unsigned)((ga >> 32) & 0x01FFFFFFu) | 0x80000000u;  // addr hi | type=2
    v8i g1;
    // data_size=1 (2B), pad_enable=1, pad_interval=3 (16 DW), pad_amount=3 (4 DW)
    g1[0] = (int)((1u << 16) | (1u << 20) | (3u << 22) | (3u << 25));
    g1[1] = (int)(((unsigned)K & 0xFFFFu) << 16);            // tensor_dim0[15:0]
    g1[2] = (int)((((unsigned)K >> 16) & 0xFFFFu) |          // tensor_dim0[31:16]
                  (((unsigned)rowsTot & 0xFFFFu) << 16));    // tensor_dim1[15:0]
    g1[3] = (int)((((unsigned)rowsTot >> 16) & 0xFFFFu) |    // tensor_dim1[31:16]
                  (32u << 16));                              // tile_dim0 = 32 elems
    g1[4] = 64;                                              // tile_dim1=64, tile_dim2=0
    g1[5] = (int)(unsigned)K;                                // tensor_dim0_stride lo32
    g1[6] = 0;
    g1[7] = 0;
    v4i z4 = {0, 0, 0, 0};
    v8i z8 = {0, 0, 0, 0, 0, 0, 0, 0};
    // 6-arg form (this toolchain): (g0, g1, g2, g3, g4, cpol)
    __builtin_amdgcn_tensor_load_to_lds(g0, g1, z4, z4, z8, 0);
}
#endif

// ---------------------------------------------------------------------------
// f32 -> bf16 conversion (grid-stride)
// ---------------------------------------------------------------------------
__global__ void f32_to_bf16(const float* __restrict__ src, bf16* __restrict__ dst,
                            size_t n) {
    size_t i = (size_t)blockIdx.x * blockDim.x + threadIdx.x;
    size_t stride = (size_t)gridDim.x * blockDim.x;
    for (; i < n; i += stride) dst[i] = (bf16)src[i];
}

// ---------------------------------------------------------------------------
// Tiled bf16 WMMA GEMM:  C[M,N] = op( A[M,K] * W[N,K]^T + bias )
// Block = 128 threads = 4 waves (2x2), block tile 64(M) x 128(N).
// Wave tile 32x64 = 2x4 fragments of v_wmma_f32_16x16x32_bf16.
// A tile is staged through LDS by the Tensor Data Mover (double-buffered);
// B (weights) streams straight from L2 via global_load_b128.
// EPI 0: relu -> bf16        (bias per n)
// EPI 1: +bias -> f32        (bias per n)
// EPI 2: per-row weight select (mask? Wmod : Wa), +biasBN[b,n], exact GELU -> bf16
// ---------------------------------------------------------------------------
template <int EPI>
__global__ __launch_bounds__(128)
void gemm_bf16(const bf16* __restrict__ A, const bf16* __restrict__ Wa,
               const bf16* __restrict__ Wmod, const int* __restrict__ mask,
               const float* __restrict__ bias, const float* __restrict__ biasBN,
               void* __restrict__ Cout, int M, int N, int K, int S_per_batch) {
    const int lane  = threadIdx.x & 31;
    const int wave  = threadIdx.x >> 5;
    const int waveM = wave >> 1, waveN = wave & 1;
    const int m_base = blockIdx.y * 64 + waveM * 32;
    const int n_base = blockIdx.x * 128 + waveN * 64;
    const int lrow  = lane & 15;
    const int lhalf = (lane >> 4) & 1;
    const int bIdx  = m_base / S_per_batch;  // 64 | S_per_batch -> uniform per block

    // W row pointers for the 4 N-fragments (optional per-hidden-row select)
    const bf16* pw[4];
#pragma unroll
    for (int j = 0; j < 4; ++j) {
        int n = n_base + j * 16 + lrow;
        const bf16* wsrc = Wa;
        if (EPI == 2) wsrc = mask[bIdx * N + n] ? Wmod : Wa;
        pw[j] = wsrc + (size_t)n * K + lhalf * 8;
    }

    const v8f vzero = {0.f, 0.f, 0.f, 0.f, 0.f, 0.f, 0.f, 0.f};
    v8f acc[2][4];
#pragma unroll
    for (int i = 0; i < 2; ++i)
#pragma unroll
        for (int j = 0; j < 4; ++j) acc[i][j] = vzero;

#if USE_TDM
    constexpr int AROWB = 80;              // padded LDS row pitch (bytes)
    constexpr int ABUF  = 64 * AROWB;      // 5120 B per buffer
    __shared__ alignas(16) unsigned char ATile[2 * ABUF];
    const unsigned ldsBase = (unsigned)(unsigned long long)(uintptr_t)(void*)&ATile[0];
    const bf16* Ablk = A + (size_t)(blockIdx.y * 64) * K;

    if (wave == 0) tdm_issue_a(Ablk, ldsBase, K, M);  // prefetch chunk 0
    int buf = 0;
    for (int k0 = 0; k0 < K; k0 += 32) {
        const bool hasNext = (k0 + 32) < K;
        if (hasNext && wave == 0)
            tdm_issue_a(Ablk + k0 + 32, ldsBase + (unsigned)((buf ^ 1) * ABUF), K, M);
        if (hasNext) __builtin_amdgcn_s_wait_tensorcnt(1);
        else         __builtin_amdgcn_s_wait_tensorcnt(0);
        __syncthreads();  // publish current A chunk to all waves

        v16bf af[2], bw[4];
#pragma unroll
        for (int i = 0; i < 2; ++i) {
            const unsigned char* rp = ATile + buf * ABUF +
                (waveM * 32 + i * 16 + lrow) * AROWB + lhalf * 16;
            af[i] = ld16((const bf16*)rp, (const bf16*)(rp + 32));
        }
#pragma unroll
        for (int j = 0; j < 4; ++j) bw[j] = ld16(pw[j] + k0, pw[j] + k0 + 16);
#pragma unroll
        for (int i = 0; i < 2; ++i)
#pragma unroll
            for (int j = 0; j < 4; ++j)
                acc[i][j] = __builtin_amdgcn_wmma_f32_16x16x32_bf16(
                    false, af[i], false, bw[j], (short)0, acc[i][j], false, false);

        __syncthreads();  // done reading buf before TDM overwrites it next iter
        buf ^= 1;
    }
#else
    // Fallback: direct per-wave global loads of A fragments
    const bf16* pa[2];
#pragma unroll
    for (int i = 0; i < 2; ++i) {
        int m = m_base + i * 16 + lrow;
        pa[i] = A + (size_t)m * K + lhalf * 8;
    }
    for (int k0 = 0; k0 < K; k0 += 32) {
        v16bf af[2], bw[4];
#pragma unroll
        for (int i = 0; i < 2; ++i) af[i] = ld16(pa[i] + k0, pa[i] + k0 + 16);
#pragma unroll
        for (int j = 0; j < 4; ++j) bw[j] = ld16(pw[j] + k0, pw[j] + k0 + 16);
#pragma unroll
        for (int i = 0; i < 2; ++i)
#pragma unroll
            for (int j = 0; j < 4; ++j)
                acc[i][j] = __builtin_amdgcn_wmma_f32_16x16x32_bf16(
                    false, af[i], false, bw[j], (short)0, acc[i][j], false, false);
    }
#endif

    // Epilogue. C 16x16 f32 layout: VGPR r, lanes0-15 -> M=r, lanes16-31 -> M=r+8.
#pragma unroll
    for (int i = 0; i < 2; ++i) {
#pragma unroll
        for (int j = 0; j < 4; ++j) {
            const int n = n_base + j * 16 + lrow;
            float bn;
            if (EPI == 2) bn = biasBN[bIdx * N + n];
            else          bn = bias[n];
#pragma unroll
            for (int r = 0; r < 8; ++r) {
                const int m = m_base + i * 16 + r + lhalf * 8;
                float v = acc[i][j][r] + bn;
                if (EPI == 0) {
                    v = v > 0.f ? v : 0.f;
                    ((bf16*)Cout)[(size_t)m * N + n] = (bf16)v;
                } else if (EPI == 1) {
                    ((float*)Cout)[(size_t)m * N + n] = v;
                } else {
                    float g = 0.5f * v * (1.f + erff(v * 0.70710678118654752f));
                    ((bf16*)Cout)[(size_t)m * N + n] = (bf16)g;
                }
            }
        }
    }
}

// ---------------------------------------------------------------------------
// Radix-select of k-th largest score per batch (4 passes, 8 bits/pass).
// Order-preserving float->uint key: larger float => larger key.
// ---------------------------------------------------------------------------
__device__ inline unsigned f2key(float f) {
    unsigned u = __float_as_uint(f);
    return (u & 0x80000000u) ? ~u : (u | 0x80000000u);
}

// state[2*b] = prefix, state[2*b+1] = remaining rank
__global__ void select_init(unsigned* state, unsigned* bins, unsigned kk) {
    int t = threadIdx.x;
    if (t < 8) { state[2 * t] = 0u; state[2 * t + 1] = kk; }
    for (int i = t; i < 8 * 256; i += blockDim.x) bins[i] = 0u;
}

__global__ void select_hist(const float* __restrict__ sc,
                            const unsigned* __restrict__ state,
                            unsigned* __restrict__ bins, int pass, int perBatch,
                            int blocksPerBatch) {
    __shared__ unsigned h[256];
    const int b = blockIdx.x / blocksPerBatch;
    const int chunk = blockIdx.x % blocksPerBatch;
    for (int i = threadIdx.x; i < 256; i += blockDim.x) h[i] = 0u;
    __syncthreads();
    const unsigned prefix = state[2 * b];
    const unsigned maskHi = (pass == 3) ? 0u : (0xFFFFFFFFu << (8 * (pass + 1)));
    const float* p = sc + (size_t)b * perBatch;
    const int per = perBatch / blocksPerBatch;
    const int beg = chunk * per;
    for (int i = beg + threadIdx.x; i < beg + per; i += blockDim.x) {
        unsigned u = f2key(p[i]);
        if ((u & maskHi) == (prefix & maskHi))
            atomicAdd(&h[(u >> (8 * pass)) & 255u], 1u);
    }
    __syncthreads();
    for (int i = threadIdx.x; i < 256; i += blockDim.x)
        if (h[i]) atomicAdd(&bins[b * 256 + i], h[i]);
}

__global__ void select_scan(unsigned* state, unsigned* bins, int pass) {
    int t = threadIdx.x;
    if (t < 8) {
        unsigned kk = state[2 * t + 1];
        unsigned prefix = state[2 * t];
        for (int v = 255; v >= 0; --v) {
            unsigned c = bins[t * 256 + v];
            if (kk <= c) { prefix |= ((unsigned)v) << (8 * pass); break; }
            kk -= c;
        }
        state[2 * t] = prefix;
        state[2 * t + 1] = kk;
    }
    __syncthreads();
    for (int i = t; i < 8 * 256; i += blockDim.x) bins[i] = 0u;  // ready for next pass
}

// mask[b,h] = OR_s (key(score[b,s,h]) >= threshold_b); bsel[b,h] = mask? mod_b : up_b
__global__ void build_mask(const float* __restrict__ sc,
                           const unsigned* __restrict__ state,
                           const float* __restrict__ up_b,
                           const float* __restrict__ mod_b,
                           int* __restrict__ maskOut, float* __restrict__ bsel,
                           int S, int H) {
    const int b = blockIdx.y;
    const int hcol = blockIdx.x * blockDim.x + threadIdx.x;
    const unsigned thr = state[2 * b];
    const float* p = sc + (size_t)b * S * H + hcol;
    int sel = 0;
    for (int s = 0; s < S; ++s) sel |= (f2key(p[(size_t)s * H]) >= thr) ? 1 : 0;
    maskOut[b * H + hcol] = sel;
    bsel[b * H + hcol] = sel ? mod_b[hcol] : up_b[hcol];
}

// ---------------------------------------------------------------------------
// Launcher
// ---------------------------------------------------------------------------
extern "C" void kernel_launch(void* const* d_in, const int* in_sizes, int n_in,
                              void* d_out, int out_size, void* d_ws, size_t ws_size,
                              hipStream_t stream) {
    const int Bt = 8, S = 256, D = 1024, H = 4096;
    const int M = Bt * S;                       // 2048 tokens
    const unsigned kSel = 1024u * 256u;         // TOP_K * S = 262144

    const float* in_x  = (const float*)d_in[0];
    const float* up_w  = (const float*)d_in[1];
    const float* up_b  = (const float*)d_in[2];
    const float* mod_w = (const float*)d_in[3];
    const float* mod_b = (const float*)d_in[4];
    const float* g1w   = (const float*)d_in[5];
    const float* g1b   = (const float*)d_in[6];
    const float* g2w   = (const float*)d_in[7];
    const float* g2b   = (const float*)d_in[8];
    const float* dww   = (const float*)d_in[9];
    const float* dwb   = (const float*)d_in[10];

    char* ws = (char*)d_ws;
    size_t off = 0;
    auto alloc = [&](size_t bytes) -> char* {
        char* p = ws + off;
        off += (bytes + 255) & ~(size_t)255;
        return p;
    };

    bf16* Xb   = (bf16*)alloc((size_t)M * D * 2);   //   4 MB  bf16 inputs
    bf16* W1b  = (bf16*)alloc((size_t)H * D * 2);   //   8 MB  gate1_w
    bf16* W2b  = (bf16*)alloc((size_t)H * H * 2);   //  32 MB  gate2_w
    bf16* UPb  = (bf16*)alloc((size_t)H * D * 2);   //   8 MB  up_w
    bf16* MODb = (bf16*)alloc((size_t)H * D * 2);   //   8 MB  mod_w
    bf16* DWb  = (bf16*)alloc((size_t)D * H * 2);   //   8 MB  down_w
    bf16* Y1b  = (bf16*)alloc((size_t)M * H * 2);   //  16 MB  relu(gate1); reused as gelu(h)
    float* SC  = (float*)alloc((size_t)M * H * 4);  //  32 MB  scores f32
    unsigned* bins  = (unsigned*)alloc(8 * 256 * 4);
    unsigned* state = (unsigned*)alloc(64);
    int*   maskM = (int*)alloc((size_t)Bt * H * 4);
    float* bsel  = (float*)alloc((size_t)Bt * H * 4);

    // ---- f32 -> bf16 conversions (all weight data becomes L2-resident) ----
    f32_to_bf16<<<4096, 256, 0, stream>>>(in_x,  Xb,   (size_t)M * D);
    f32_to_bf16<<<4096, 256, 0, stream>>>(g1w,   W1b,  (size_t)H * D);
    f32_to_bf16<<<4096, 256, 0, stream>>>(g2w,   W2b,  (size_t)H * H);
    f32_to_bf16<<<4096, 256, 0, stream>>>(up_w,  UPb,  (size_t)H * D);
    f32_to_bf16<<<4096, 256, 0, stream>>>(mod_w, MODb, (size_t)H * D);
    f32_to_bf16<<<4096, 256, 0, stream>>>(dww,   DWb,  (size_t)D * H);

    const dim3 blk(128);
    const dim3 gGate(H / 128, M / 64);  // (32, 32)

    // gate1: Y1 = relu(X*W1^T + b1)  -> bf16
    gemm_bf16<0><<<gGate, blk, 0, stream>>>(Xb, W1b, nullptr, nullptr, g1b,
                                            nullptr, Y1b, M, H, D, S);
    // gate2: SC = Y1*W2^T + b2  -> f32
    gemm_bf16<1><<<gGate, blk, 0, stream>>>(Y1b, W2b, nullptr, nullptr, g2b,
                                            nullptr, SC, M, H, H, S);

    // ---- per-batch k-th-largest threshold via radix select ----
    select_init<<<1, 256, 0, stream>>>(state, bins, kSel);
    for (int pass = 3; pass >= 0; --pass) {
        select_hist<<<8 * 64, 256, 0, stream>>>(SC, state, bins, pass, S * H, 64);
        select_scan<<<1, 256, 0, stream>>>(state, bins, pass);
    }
    build_mask<<<dim3(H / 256, Bt), 256, 0, stream>>>(SC, state, up_b, mod_b,
                                                      maskM, bsel, S, H);

    // h = gelu(X * Wsel^T + bsel)  (per-hidden-row select mod_w/up_w) -> bf16 (reuse Y1b)
    gemm_bf16<2><<<gGate, blk, 0, stream>>>(Xb, UPb, MODb, maskM, nullptr, bsel,
                                            Y1b, M, H, D, S);

    // out = h * down_w^T + down_b -> f32
    const dim3 gDown(D / 128, M / 64);  // (8, 32)
    gemm_bf16<1><<<gDown, blk, 0, stream>>>(Y1b, DWb, nullptr, nullptr, dwb,
                                            nullptr, d_out, M, D, H, S);
}
